// MLKVFullAttnDecodeStep_10969346474367
// MI455X (gfx1250) — matrix operand order repaired and checked by reading, not compiled, not run on hardware
//
#include <hip/hip_runtime.h>
#include <hip/hip_bf16.h>
#include <math.h>

typedef __attribute__((ext_vector_type(16))) _Float16 v16h;
typedef __attribute__((ext_vector_type(8)))  float    v8f;

#define H_   32
#define HKV_ 8
#define D_   128
#define HID_ 2048
#define S_   8192
#define RD_  32
#define G_   4
#define KIDX_ 2

// ---- workspace layout (floats) ----
#define OFF_Y   0            // qg[8192] | kvec[1024] | vvec[1024]
#define OFF_QF  10240        // q_final[32*128]
#define OFF_KN  14336        // k_new[8*128]
#define OFF_PR  16384        // scores/probs [32 * 8192]
#define OFF_OP  278528       // out partials [8 segs * 4096]
#define OFF_OV  311296       // gated out vec [4096]

// ---------------- GEMV: y = W @ x  (row-major W, rows x cols) ----------------
__global__ void __launch_bounds__(256) gemv_kernel(const float* __restrict__ W,
                                                   const float* __restrict__ x,
                                                   float* __restrict__ y,
                                                   int rows, int cols) {
    extern __shared__ float sx[];
    const int tid = threadIdx.x;
    for (int i = tid; i < cols; i += 256) sx[i] = x[i];
    __syncthreads();
    const int wave = tid >> 5, lane = tid & 31;
    const int row = blockIdx.x * 8 + wave;
    if (row >= rows) return;
    const float4* Wr = (const float4*)(W + (size_t)row * cols);
    const float4* x4 = (const float4*)sx;
    const int n4 = cols >> 2;
    float acc = 0.f;
    for (int i = lane; i < n4; i += 32) {
        float4 w = Wr[i], v = x4[i];
        acc += w.x*v.x + w.y*v.y + w.z*v.z + w.w*v.w;
    }
    #pragma unroll
    for (int off = 16; off > 0; off >>= 1) acc += __shfl_down(acc, off, 32);
    if (lane == 0) y[row] = acc;
}

// ---------------- prep: RMS-norm + RoPE for q (32 heads) and k (8 heads) ----------------
__global__ void __launch_bounds__(128) prep_kernel(const float* __restrict__ qg,
                                                   const float* __restrict__ kvec,
                                                   const float* __restrict__ cosv,
                                                   const float* __restrict__ sinv,
                                                   const float* __restrict__ qnw,
                                                   const float* __restrict__ knw,
                                                   float* __restrict__ q_final,
                                                   float* __restrict__ k_new) {
    __shared__ float red[128];
    __shared__ float vals[128];
    const int b = blockIdx.x, tid = threadIdx.x;
    const float* src; const float* w; float* dst;
    if (b < 32) { src = qg + (size_t)b * 256; w = qnw; dst = q_final + (size_t)b * 128; }
    else { int h = b - 32; src = kvec + (size_t)h * 128; w = knw; dst = k_new + (size_t)h * 128; }
    float v = src[tid];
    red[tid] = v * v;
    __syncthreads();
    #pragma unroll
    for (int o = 64; o > 0; o >>= 1) { if (tid < o) red[tid] += red[tid + o]; __syncthreads(); }
    float rms = rsqrtf(red[0] * (1.f / 128.f) + 1e-6f);
    float vn = v * rms * (1.f + w[tid]);
    vals[tid] = vn;
    __syncthreads();
    float out;
    if (tid < 16)      out = vn * cosv[tid] - vals[tid + 16] * sinv[tid];
    else if (tid < 32) out = vn * cosv[tid] + vals[tid - 16] * sinv[tid];
    else               out = vn;
    dst[tid] = out;
}

// ---------------- scores: per (head, 64-key chunk), one wave, WMMA f16 ----------------
// D = A(16 keys x 32 dims) x B(32 dims x 16 queries), accumulated over 4 d-slices.
__global__ void __launch_bounds__(32) scores_kernel(const float* __restrict__ q_final,
                                                    const float* __restrict__ k_new,
                                                    const float* __restrict__ kv_cache,
                                                    const int* __restrict__ posp,
                                                    float* __restrict__ probs) {
    const int pos = *posp;
    const int bid = blockIdx.x;
    const int h = bid >> 7;
    const int chunk = bid & 127;
    const int s0 = chunk * 64;
    if (s0 > pos) return;
    const int lane = threadIdx.x;
    const int half = lane >> 4;
    const int l15 = lane & 15;

    // B tiles: Q^T, lane = query column, K = head-dim slice
    const int q = l15;
    const float qm = (q < 4) ? 1.f : 0.f;
    const float* qrow = q_final + (size_t)(h * 4 + ((q < 4) ? q : 0)) * 128;
    v16h b[4];
    #pragma unroll
    for (int ds = 0; ds < 4; ++ds) {
        const int dbase = ds * 32 + half * 8;
        v16h t;
        #pragma unroll
        for (int i = 0; i < 16; ++i) t[i] = (_Float16)(qrow[dbase + i + ((i >= 8) ? 8 : 0)] * qm);
        b[ds] = t;
    }

    const float* Kbase = kv_cache + (size_t)(KIDX_ * HKV_ + h) * S_ * D_;
    const float scale = 0.08838834764831845f; // 1/sqrt(128)

    #pragma unroll
    for (int t = 0; t < 4; ++t) {
        const int s = s0 + t * 16 + l15;
        const float* krow = (s == pos) ? (k_new + (size_t)h * 128)
                                       : (Kbase + (size_t)s * D_);
        v8f c = {};
        #pragma unroll
        for (int ds = 0; ds < 4; ++ds) {
            const int dOff = ds * 32 + half * 8;
            float4 w0 = *(const float4*)(krow + dOff);
            float4 w1 = *(const float4*)(krow + dOff + 4);
            float4 w2 = *(const float4*)(krow + dOff + 16);
            float4 w3 = *(const float4*)(krow + dOff + 20);
            v16h a;
            a[0]=(_Float16)w0.x; a[1]=(_Float16)w0.y; a[2]=(_Float16)w0.z; a[3]=(_Float16)w0.w;
            a[4]=(_Float16)w1.x; a[5]=(_Float16)w1.y; a[6]=(_Float16)w1.z; a[7]=(_Float16)w1.w;
            a[8]=(_Float16)w2.x; a[9]=(_Float16)w2.y; a[10]=(_Float16)w2.z; a[11]=(_Float16)w2.w;
            a[12]=(_Float16)w3.x; a[13]=(_Float16)w3.y; a[14]=(_Float16)w3.z; a[15]=(_Float16)w3.w;
            c = __builtin_amdgcn_wmma_f32_16x16x32_f16(false, a, false, b[ds],
                                                       (short)0, c, false, false);
        }
        if (q < 4) {
            const size_t base = (size_t)(h * 4 + q) * S_ + s0 + t * 16 + half * 8;
            #pragma unroll
            for (int r = 0; r < 8; ++r) probs[base + r] = c[r] * scale;
        }
    }
}

// ---------------- softmax per (h,g) row over s <= pos; zero-pad to 64-multiple ----------------
__global__ void __launch_bounds__(256) softmax_kernel(float* __restrict__ probs,
                                                      const int* __restrict__ posp) {
    __shared__ float red[256];
    const int pos = *posp;
    const int tid = threadIdx.x;
    float* sc = probs + (size_t)blockIdx.x * S_;
    float m = -1e30f;
    for (int s = tid; s <= pos; s += 256) m = fmaxf(m, sc[s]);
    red[tid] = m;
    __syncthreads();
    #pragma unroll
    for (int o = 128; o > 0; o >>= 1) { if (tid < o) red[tid] = fmaxf(red[tid], red[tid + o]); __syncthreads(); }
    m = red[0];
    __syncthreads();
    float sum = 0.f;
    for (int s = tid; s <= pos; s += 256) sum += expf(sc[s] - m);
    red[tid] = sum;
    __syncthreads();
    #pragma unroll
    for (int o = 128; o > 0; o >>= 1) { if (tid < o) red[tid] += red[tid + o]; __syncthreads(); }
    const float inv = 1.f / red[0];
    const int nR = (pos + 64) & ~63;   // round_up(pos+1, 64)
    for (int s = tid; s < nR; s += 256)
        sc[s] = (s <= pos) ? expf(sc[s] - m) * inv : 0.f;
}

// ---------------- attn @ V: per (head, dtile, s-segment), one wave, WMMA f16 ----------------
// D = A(16 queries x 32 keys) x B(32 keys x 16 dims), accumulated over key chunks.
__global__ void __launch_bounds__(32) attnv_kernel(const float* __restrict__ probs,
                                                   const float* __restrict__ kv_cache,
                                                   const float* __restrict__ vvec,
                                                   const int* __restrict__ posp,
                                                   float* __restrict__ out_part) {
    const int pos = *posp;
    const int bid = blockIdx.x;          // h*64 + dtile*8 + seg
    const int h = bid >> 6;
    const int dtile = (bid >> 3) & 7;
    const int seg = bid & 7;
    const int lane = threadIdx.x;
    const int half = lane >> 4;
    const int l15 = lane & 15;
    const float* Vbase = kv_cache + (size_t)((KIDX_ + 1) * HKV_ + h) * S_ * D_;
    const int d = dtile * 16 + l15;
    const int q = l15;
    const float qm = (q < 4) ? 1.f : 0.f;
    const float* prow = probs + (size_t)(h * 4 + ((q < 4) ? q : 0)) * S_;

    v8f c = {};
    const int segEnd0 = seg * 1024 + 1024;
    const int sEnd = (segEnd0 < pos + 1) ? segEnd0 : (pos + 1);
    for (int s0 = seg * 1024; s0 < sEnd; s0 += 32) {
        // A: P tile (queries x keys)
        v16h a;
        const int k0 = s0 + half * 8;
        #pragma unroll
        for (int i = 0; i < 8; ++i) a[i]     = (_Float16)(prow[k0 + i] * qm);
        #pragma unroll
        for (int i = 0; i < 8; ++i) a[8 + i] = (_Float16)(prow[k0 + 16 + i] * qm);
        // B: V tile (keys x dims), column d per lane
        v16h b;
        #pragma unroll
        for (int i = 0; i < 16; ++i) {
            const int s = s0 + half * 8 + i + ((i >= 8) ? 8 : 0);
            const float* vp = (s == pos) ? (vvec + (size_t)h * 128 + d)
                                         : (Vbase + (size_t)s * D_ + d);
            b[i] = (_Float16)(*vp);
        }
        c = __builtin_amdgcn_wmma_f32_16x16x32_f16(false, a, false, b,
                                                   (short)0, c, false, false);
    }
    if (half == 0) {
        #pragma unroll
        for (int r = 0; r < 4; ++r)
            out_part[(size_t)seg * 4096 + (size_t)(h * 4 + r) * 128 + d] = c[r];
    }
}

// ---------------- reduce segment partials + sigmoid gate ----------------
__global__ void __launch_bounds__(256) reduce_gate_kernel(const float* __restrict__ out_part,
                                                          const float* __restrict__ qg,
                                                          float* __restrict__ out_vec) {
    const int i = blockIdx.x * 256 + threadIdx.x;   // 0..4095
    float sum = 0.f;
    #pragma unroll
    for (int seg = 0; seg < 8; ++seg) sum += out_part[(size_t)seg * 4096 + i];
    const int hfull = i >> 7, d = i & 127;
    const float g = qg[(size_t)hfull * 256 + 128 + d];
    out_vec[i] = sum / (1.f + expf(-g));
}

extern "C" void kernel_launch(void* const* d_in, const int* in_sizes, int n_in,
                              void* d_out, int out_size, void* d_ws, size_t ws_size,
                              hipStream_t stream) {
    const float* hidden   = (const float*)d_in[0];
    const float* cosv     = (const float*)d_in[1];
    const float* sinv     = (const float*)d_in[2];
    // d_in[3] causal_mask unused: s<=pos restriction is exact (exp(-1e9) == 0 in fp32)
    const float* kv_cache = (const float*)d_in[4];
    const float* wq       = (const float*)d_in[5];
    const float* wk       = (const float*)d_in[6];
    const float* wv       = (const float*)d_in[7];
    const float* wo       = (const float*)d_in[8];
    const float* qnw      = (const float*)d_in[9];
    const float* knw      = (const float*)d_in[10];
    const int*   posp     = (const int*)d_in[11];

    float* ws   = (float*)d_ws;
    float* qg   = ws + OFF_Y;          // 8192
    float* kvec = ws + OFF_Y + 8192;   // 1024
    float* vvec = ws + OFF_Y + 9216;   // 1024
    float* qf   = ws + OFF_QF;
    float* kn   = ws + OFF_KN;
    float* pr   = ws + OFF_PR;
    float* opar = ws + OFF_OP;
    float* ovec = ws + OFF_OV;
    float* out  = (float*)d_out;

    // 1) QKV projections (bandwidth-dominant: 80 MB of weights)
    gemv_kernel<<<1024, 256, HID_ * sizeof(float), stream>>>(wq, hidden, qg,  8192, HID_);
    gemv_kernel<<< 128, 256, HID_ * sizeof(float), stream>>>(wk, hidden, kvec, 1024, HID_);
    gemv_kernel<<< 128, 256, HID_ * sizeof(float), stream>>>(wv, hidden, vvec, 1024, HID_);
    // 2) RMS-norm + RoPE
    prep_kernel<<<40, 128, 0, stream>>>(qg, kvec, cosv, sinv, qnw, knw, qf, kn);
    // 3) scores (WMMA), 8 heads x 128 key-chunks
    scores_kernel<<<1024, 32, 0, stream>>>(qf, kn, kv_cache, posp, pr);
    // 4) softmax per (h,g)
    softmax_kernel<<<32, 256, 0, stream>>>(pr, posp);
    // 5) attn @ V (WMMA), split over 8 s-segments for bandwidth
    attnv_kernel<<<512, 32, 0, stream>>>(pr, kv_cache, vvec, posp, opar);
    // 6) combine + gate
    reduce_gate_kernel<<<16, 256, 0, stream>>>(opar, qg, ovec);
    // 7) output projection (32 MB of weights)
    gemv_kernel<<<256, 256, (H_ * D_) * sizeof(float), stream>>>(wo, ovec, out, HID_, H_ * D_);
}
